// ODModel_89421219103305
// MI455X (gfx1250) — matrix operand order repaired and checked by reading, not compiled, not run on hardware
//
#include <hip/hip_runtime.h>

typedef __attribute__((ext_vector_type(16))) _Float16 v16h;
typedef __attribute__((ext_vector_type(8)))  _Float16 v8h;
typedef __attribute__((ext_vector_type(8)))  float    v8f;

#define BN_EPS 1e-5f

static __device__ inline v16h cat8(v8h l, v8h h) {
    return __builtin_shufflevector(l, h, 0,1,2,3,4,5,6,7,8,9,10,11,12,13,14,15);
}
static __device__ inline v16h ldA(const _Float16* arow, int ci, int kbA) {
    return cat8(*(const v8h*)(arow + ci + kbA),
                *(const v8h*)(arow + ci + kbA + 8));
}
static __device__ inline v16h ldB(const _Float16* base, int ci, int kbB) {
    return cat8(*(const v8h*)(base + ci + kbB),
                *(const v8h*)(base + ci + kbB + 8));
}

struct Frags { v16h a0, a1, b0, b1, b2, b3; };

static __device__ inline Frags ldstep(const _Float16* pa0, const _Float16* pa1,
                                      const _Float16* pb0, const _Float16* pb1,
                                      const _Float16* pb2, const _Float16* pb3,
                                      int ci, int kbA, int kbB)
{
    Frags f;
    f.a0 = ldA(pa0, ci, kbA);
    f.a1 = ldA(pa1, ci, kbA);
    f.b0 = ldB(pb0, ci, kbB);
    f.b1 = ldB(pb1, ci, kbB);
    f.b2 = ldB(pb2, ci, kbB);
    f.b3 = ldB(pb3, ci, kbB);
    return f;
}

static __device__ inline void mstep(const Frags& f, v8f acc0[4], v8f acc1[4])
{
    acc0[0] = __builtin_amdgcn_wmma_f32_16x16x32_f16(false, f.a0, false, f.b0, (short)0, acc0[0], false, false);
    acc1[0] = __builtin_amdgcn_wmma_f32_16x16x32_f16(false, f.a1, false, f.b0, (short)0, acc1[0], false, false);
    acc0[1] = __builtin_amdgcn_wmma_f32_16x16x32_f16(false, f.a0, false, f.b1, (short)0, acc0[1], false, false);
    acc1[1] = __builtin_amdgcn_wmma_f32_16x16x32_f16(false, f.a1, false, f.b1, (short)0, acc1[1], false, false);
    acc0[2] = __builtin_amdgcn_wmma_f32_16x16x32_f16(false, f.a0, false, f.b2, (short)0, acc0[2], false, false);
    acc1[2] = __builtin_amdgcn_wmma_f32_16x16x32_f16(false, f.a1, false, f.b2, (short)0, acc1[2], false, false);
    acc0[3] = __builtin_amdgcn_wmma_f32_16x16x32_f16(false, f.a0, false, f.b3, (short)0, acc0[3], false, false);
    acc1[3] = __builtin_amdgcn_wmma_f32_16x16x32_f16(false, f.a1, false, f.b3, (short)0, acc1[3], false, false);
}

// ---------------------------------------------------------------------------
// Channels-last f16 implicit-GEMM conv via V_WMMA_F32_16X16X32_F16.
// One wave32 computes a 32 (co) x 64 (pixel) tile: 8 accumulators.
// K is processed in 64-half chunks (2 WMMA K-steps, immediate-offset b128
// loads); the next chunk's 24 loads are issued before the current chunk's
// 16 WMMAs so loads stay in flight under the matrix ops, with bounded
// register pressure (no spills, no VGPR-MSB traffic).
// X  : f16 [p][CIN], p = n*H*W + h*W + w
// W16: f16 [co][K],  K-order = (r*KS+s)*CIN + ci  (matches B gather order)
// Y  : f16 [p][ldY]
// zb : >= 2048 zero halves (pointer target for OOB rows/cols)
// mode 0: fused BN + LeakyReLU(0.1); mode 1: +bias (detect)
// ---------------------------------------------------------------------------
template<int CIN, int KS>
__global__ __launch_bounds__(256) void conv_cl_wmma(
    const _Float16* __restrict__ X, const _Float16* __restrict__ W16,
    const float* __restrict__ g, const float* __restrict__ bnb,
    const float* __restrict__ m, const float* __restrict__ v,
    const float* __restrict__ bias,
    _Float16* __restrict__ Y, const _Float16* __restrict__ zb,
    int Bn, int Cout, int H, int Wd, int ldY, int mode)
{
    const int HW  = H * Wd;
    const int NP  = Bn * HW;
    const int npt = (NP + 63) >> 6;          // 64-pixel tiles
    const int nct = (Cout + 31) >> 5;        // 32-channel tiles

    const int wave = (blockIdx.x << 3) + (threadIdx.x >> 5);
    if (wave >= npt * nct) return;           // wave-uniform
    const int ct = wave % nct;               // waves in a block share pt
    const int pt = wave / nct;

    const int lane = threadIdx.x & 31;
    const int lo   = lane & 15;
    const int hi   = lane >> 4;
    const int kbA  = hi ? 16 : 0;            // A lane K-offset
    const int kbB  = hi ? 16 : 0;            // B lane K-offset

    // A rows (weights) for this lane: two co sub-tiles
    const int  K = CIN * KS * KS;
    const int  co_a0 = ct * 32 + lo;
    const int  co_a1 = co_a0 + 16;
    const bool cv0 = co_a0 < Cout;
    const bool cv1 = co_a1 < Cout;
    const _Float16* wrow0 = cv0 ? (W16 + (long long)co_a0 * K) : zb;
    const _Float16* wrow1 = cv1 ? (W16 + (long long)co_a1 * K) : zb;

    // B columns: 4 pixel sub-tiles
    int  nT[4], hT[4], wT[4];
    bool vT[4];
    #pragma unroll
    for (int t = 0; t < 4; ++t) {
        const int p = pt * 64 + t * 16 + lo;
        vT[t] = p < NP;
        const int pc = vT[t] ? p : 0;
        nT[t] = pc / HW;
        const int hw = pc - nT[t] * HW;
        hT[t] = hw / Wd;
        wT[t] = hw - hT[t] * Wd;
    }

    constexpr int pad = (KS - 1) >> 1;
    v8f acc0[4], acc1[4];
    #pragma unroll
    for (int t = 0; t < 4; ++t) {
        acc0[t] = (v8f){0.f,0.f,0.f,0.f,0.f,0.f,0.f,0.f};
        acc1[t] = (v8f){0.f,0.f,0.f,0.f,0.f,0.f,0.f,0.f};
    }

    #pragma unroll 1
    for (int rs = 0; rs < KS * KS; ++rs) {
        const int r = rs / KS;
        const int s = rs - r * KS;
        const _Float16* pa0 = cv0 ? (wrow0 + (long long)rs * CIN) : zb;
        const _Float16* pa1 = cv1 ? (wrow1 + (long long)rs * CIN) : zb;
        const _Float16* pb[4];
        #pragma unroll
        for (int t = 0; t < 4; ++t) {
            const int hh = hT[t] + r - pad;
            const int ww = wT[t] + s - pad;
            const bool ok = vT[t] && ((unsigned)hh < (unsigned)H) &&
                                     ((unsigned)ww < (unsigned)Wd);
            pb[t] = ok ? (X + (long long)(nT[t] * HW + hh * Wd + ww) * CIN) : zb;
        }
        const _Float16* pb0 = pb[0];
        const _Float16* pb1 = pb[1];
        const _Float16* pb2 = pb[2];
        const _Float16* pb3 = pb[3];

        // prologue: chunk 0 (two K-steps)
        Frags F0 = ldstep(pa0, pa1, pb0, pb1, pb2, pb3,  0, kbA, kbB);
        Frags F1 = ldstep(pa0, pa1, pb0, pb1, pb2, pb3, 32, kbA, kbB);

        #pragma unroll 1
        for (int kc = 0; kc < CIN; kc += 64) {
            pa0 += 64; pa1 += 64;
            pb0 += 64; pb1 += 64; pb2 += 64; pb3 += 64;
            // prefetch next chunk (last iteration reads into padded slack; dead)
            const Frags G0 = ldstep(pa0, pa1, pb0, pb1, pb2, pb3,  0, kbA, kbB);
            const Frags G1 = ldstep(pa0, pa1, pb0, pb1, pb2, pb3, 32, kbA, kbB);
            mstep(F0, acc0, acc1);
            mstep(F1, acc0, acc1);
            F0 = G0; F1 = G1;
        }
    }

    __builtin_amdgcn_sched_barrier(0);       // keep epilogue out of the hot loop

    // ---- epilogue: accU[t][i] is (co = ct*32 + u*16 + i + 8*hi,
    //                               pixel = pt*64 + t*16 + lo) -----------------
    float sc[2][8], sh[2][8];
    #pragma unroll
    for (int u = 0; u < 2; ++u) {
        #pragma unroll
        for (int i = 0; i < 8; ++i) {
            const int co = ct * 32 + u * 16 + i + 8 * hi;
            const int cc = co < Cout ? co : 0;
            if (mode == 0) {
                sc[u][i] = g[cc] * rsqrtf(v[cc] + BN_EPS);
                sh[u][i] = bnb[cc] - m[cc] * sc[u][i];
            } else {
                sc[u][i] = 1.f;
                sh[u][i] = bias[cc];
            }
        }
    }
    #pragma unroll
    for (int t = 0; t < 4; ++t) {
        const int p = pt * 64 + t * 16 + lo;
        if (p >= NP) continue;
        #pragma unroll
        for (int u = 0; u < 2; ++u) {
            const int cbase = ct * 32 + u * 16 + 8 * hi;
            if (cbase >= Cout) continue;   // detect tail tiles
            v8h o;
            #pragma unroll
            for (int i = 0; i < 8; ++i) {
                const float av = (u == 0) ? acc0[t][i] : acc1[t][i];
                float r = av * sc[u][i] + sh[u][i];
                if (mode == 0) r = (r > 0.f) ? r : 0.1f * r;
                o[i] = (_Float16)r;
            }
            *(v8h*)(Y + (long long)p * ldY + cbase) = o;
        }
    }
}

// ---------------------------------------------------------------------------
// Weight repack: f32 (Cout,Cin,ks,ks) -> f16 [co][rs*Cin+ci]
// ---------------------------------------------------------------------------
__global__ void repack_w16(const float* __restrict__ w, _Float16* __restrict__ out,
                           int Cout, int Cin, int ksks)
{
    const long long idx = (long long)blockIdx.x * blockDim.x + threadIdx.x;
    const long long total = (long long)Cout * Cin * ksks;
    if (idx >= total) return;
    const int rs = (int)(idx % ksks);
    long long t  = idx / ksks;
    const int ci = (int)(t % Cin);
    const int co = (int)(t / Cin);
    out[(long long)co * Cin * ksks + (long long)rs * Cin + ci] = (_Float16)w[idx];
}

__global__ void zero16_kernel(_Float16* z, int n)
{
    const int i = blockIdx.x * blockDim.x + threadIdx.x;
    if (i < n) z[i] = (_Float16)0.f;
}

// NCHW f32 -> channels-last f16
__global__ void nchw_to_cl16(const float* __restrict__ X, _Float16* __restrict__ Y,
                             int Bn, int C, int HW)
{
    const long long idx = (long long)blockIdx.x * blockDim.x + threadIdx.x;
    const long long total = (long long)Bn * C * HW;
    if (idx >= total) return;
    const int c = (int)(idx % C);
    long long t = idx / C;
    const int hw = (int)(t % HW);
    const int n  = (int)(t / HW);
    Y[idx] = (_Float16)X[((long long)n * C + c) * HW + hw];
}

// 2x nearest upsample of U (f16 CL, half-res, C1 ch) ++ F (f32 NCHW, C2 ch)
// -> Y f16 CL (B,H,W,C1+C2)
__global__ void upcat_cl16(const _Float16* __restrict__ U, const float* __restrict__ F,
                           _Float16* __restrict__ Y, int Bn, int C1, int C2,
                           int H, int W)
{
    const int C = C1 + C2;
    const long long idx = (long long)blockIdx.x * blockDim.x + threadIdx.x;
    const long long total = (long long)Bn * H * W * C;
    if (idx >= total) return;
    const int c = (int)(idx % C);
    long long t = idx / C;
    const int w = (int)(t % W); t /= W;
    const int h = (int)(t % H);
    const int n = (int)(t / H);
    _Float16 val;
    if (c < C1) {
        const long long ph = (long long)(n * (H >> 1) + (h >> 1)) * (W >> 1) + (w >> 1);
        val = U[ph * C1 + c];
    } else {
        val = (_Float16)F[(((long long)n * C2 + (c - C1)) * H + h) * W + w];
    }
    Y[idx] = val;
}

// ---------------------------------------------------------------------------
// Decode: P f16 CL [p][80] (c = a*25+e contiguous) -> f32 rows of OUT(B,10647,25)
// ---------------------------------------------------------------------------
__global__ void decode_kernel(const _Float16* __restrict__ P, float* __restrict__ OUT,
                              int Bn, int G, int rowOff, float stride,
                              float aw0, float ah0, float aw1, float ah1,
                              float aw2, float ah2)
{
    const int idx = blockIdx.x * blockDim.x + threadIdx.x;
    const int total = Bn * 3 * G * G;
    if (idx >= total) return;
    const int gx = idx % G;
    int t = idx / G;
    const int gy = t % G; t /= G;
    const int a  = t % 3;
    const int n  = t / 3;

    const float aw = (a == 0) ? aw0 : ((a == 1) ? aw1 : aw2);
    const float ah = (a == 0) ? ah0 : ((a == 1) ? ah1 : ah2);

    const long long pix = (long long)n * G * G + (long long)gy * G + gx;
    const _Float16* p = P + pix * 80 + a * 25;
    float* o = OUT + ((long long)n * 10647 + rowOff
                      + (long long)a * G * G + (long long)gy * G + gx) * 25;

    const float t0 = (float)p[0], t1 = (float)p[1], t2 = (float)p[2], t3 = (float)p[3];
    o[0] = (1.f / (1.f + __expf(-t0)) + (float)gx) * stride;
    o[1] = (1.f / (1.f + __expf(-t1)) + (float)gy) * stride;
    o[2] = __expf(t2) * aw;
    o[3] = __expf(t3) * ah;
    #pragma unroll
    for (int e = 4; e < 25; ++e)
        o[e] = 1.f / (1.f + __expf(-(float)p[e]));
}

// ---------------------------------------------------------------------------
// Host-side orchestration
// ---------------------------------------------------------------------------
struct CB { const float *b, *g, *m, *v, *w; };

static inline CB getCB(void* const* d, int base) {
    CB c;
    c.b = (const float*)d[base + 0];
    c.g = (const float*)d[base + 1];
    c.m = (const float*)d[base + 2];
    c.v = (const float*)d[base + 3];
    c.w = (const float*)d[base + 4];
    return c;
}

// runtime (Cin,ks) -> template dispatch
static void run_conv(hipStream_t s, const _Float16* X, const _Float16* W16,
                     const float* g, const float* b, const float* m, const float* v,
                     const float* bias, _Float16* Y, const _Float16* zb,
                     int Bn, int Cin, int Cout, int H, int W, int ks,
                     int ldY, int mode)
{
    const int NP = Bn * H * W;
    const int blocks = (((NP + 63) / 64) * ((Cout + 31) / 32) + 7) / 8;
#define CONV_CASE(CI, KK)                                                      \
    if (Cin == CI && ks == KK) {                                               \
        conv_cl_wmma<CI, KK><<<blocks, 256, 0, s>>>(X, W16, g, b, m, v, bias,  \
                                                    Y, zb, Bn, Cout, H, W,     \
                                                    ldY, mode);                \
        return;                                                                \
    }
    CONV_CASE(1024, 1)
    CONV_CASE( 768, 1)
    CONV_CASE( 512, 1)
    CONV_CASE( 512, 3)
    CONV_CASE( 384, 1)
    CONV_CASE( 256, 1)
    CONV_CASE( 256, 3)
    CONV_CASE( 128, 1)
    CONV_CASE( 128, 3)
#undef CONV_CASE
}

extern "C" void kernel_launch(void* const* d_in, const int* in_sizes, int n_in,
                              void* d_out, int out_size, void* d_ws, size_t ws_size,
                              hipStream_t stream) {
    (void)in_sizes; (void)n_in; (void)out_size; (void)ws_size;
    const int Bn = 16;

    // ---- inputs (jax pytree / alphabetical dict flattening) ---------------
    const float* f13 = (const float*)d_in[0];
    const float* f26 = (const float*)d_in[1];
    const float* f52 = (const float*)d_in[2];
    const float* d1_bias = (const float*)d_in[3];
    const float* d1_w    = (const float*)d_in[4];
    CB d1cb = getCB(d_in, 5);
    const float* d2_bias = (const float*)d_in[10];
    const float* d2_w    = (const float*)d_in[11];
    CB d2cb = getCB(d_in, 12);
    const float* d3_bias = (const float*)d_in[17];
    const float* d3_w    = (const float*)d_in[18];
    CB d3cb = getCB(d_in, 19);
    CB s1[5], s2[5], s3[5];
    for (int i = 0; i < 5; ++i) s1[i] = getCB(d_in, 24 + 5 * i);
    for (int i = 0; i < 5; ++i) s2[i] = getCB(d_in, 49 + 5 * i);
    for (int i = 0; i < 5; ++i) s3[i] = getCB(d_in, 74 + 5 * i);
    CB u1 = getCB(d_in, 99);
    CB u2 = getCB(d_in, 104);

    // ---- workspace layout --------------------------------------------------
    char* ws = (char*)d_ws;
    const size_t MiB = 1024ull * 1024ull;
    _Float16* zbuf  = (_Float16*)(ws);               //  4 KB zeros
    _Float16* wbase = (_Float16*)(ws + 1   * MiB);   // 48 MiB repacked weights
    _Float16* actA  = (_Float16*)(ws + 49  * MiB);   // 12 MiB
    _Float16* actB  = (_Float16*)(ws + 61  * MiB);   // 24 MiB
    _Float16* actC  = (_Float16*)(ws + 85  * MiB);   // 34 MiB
    _Float16* actD  = (_Float16*)(ws + 119 * MiB);   //  8 MiB (detect, ld=80)
    float* out = (float*)d_out;

    zero16_kernel<<<8, 256, 0, stream>>>(zbuf, 2048);

    // ---- weight repacks ----------------------------------------------------
    size_t woff = 0;
    auto repack = [&](const float* w, int Cout, int Cin, int ks) -> _Float16* {
        _Float16* dst = wbase + woff;
        const long long tot = (long long)Cout * Cin * ks * ks;
        woff += (size_t)((tot + 255) & ~255ll);
        repack_w16<<<(int)((tot + 255) / 256), 256, 0, stream>>>(w, dst, Cout, Cin, ks * ks);
        return dst;
    };
    _Float16 *ws1[5], *ws2[5], *ws3[5];
    const int s1c[5][3] = {{512,1024,1},{1024,512,3},{512,1024,1},{1024,512,3},{512,1024,1}};
    const int s2c[5][3] = {{256, 768,1},{ 512,256,3},{256, 512,1},{ 512,256,3},{256, 512,1}};
    const int s3c[5][3] = {{128, 384,1},{ 256,128,3},{128, 256,1},{ 256,128,3},{128, 256,1}};
    for (int i = 0; i < 5; ++i) ws1[i] = repack(s1[i].w, s1c[i][0], s1c[i][1], s1c[i][2]);
    for (int i = 0; i < 5; ++i) ws2[i] = repack(s2[i].w, s2c[i][0], s2c[i][1], s2c[i][2]);
    for (int i = 0; i < 5; ++i) ws3[i] = repack(s3[i].w, s3c[i][0], s3c[i][1], s3c[i][2]);
    _Float16* wd1cb = repack(d1cb.w, 1024, 512, 3);
    _Float16* wd2cb = repack(d2cb.w,  512, 256, 3);
    _Float16* wd3cb = repack(d3cb.w,  256, 128, 3);
    _Float16* wu1   = repack(u1.w, 256, 512, 1);
    _Float16* wu2   = repack(u2.w, 128, 256, 1);
    _Float16* wd1   = repack(d1_w, 75, 1024, 1);
    _Float16* wd2   = repack(d2_w, 75,  512, 1);
    _Float16* wd3   = repack(d3_w, 75,  256, 1);

    // ---- conv launch helpers ----------------------------------------------
    auto convCB = [&](const _Float16* X, const CB& p, const _Float16* W16,
                      _Float16* Y, int Cin, int Cout, int H, int W, int ks) {
        run_conv(stream, X, W16, p.g, p.b, p.m, p.v, nullptr, Y, zbuf,
                 Bn, Cin, Cout, H, W, ks, Cout, 0);
    };
    auto convDet = [&](const _Float16* X, const _Float16* W16, const float* bias,
                       _Float16* Y, int Cin, int H, int W) {
        run_conv(stream, X, W16, nullptr, nullptr, nullptr, nullptr, bias, Y, zbuf,
                 Bn, Cin, 75, H, W, 1, 80, 1);
    };

    // ================= scale 1 : 13x13 =================
    {
        const long long tot = (long long)Bn * 1024 * 169;
        nchw_to_cl16<<<(int)((tot + 255) / 256), 256, 0, stream>>>(f13, actC, Bn, 1024, 169);
    }
    convCB(actC, s1[0], ws1[0], actA, 1024,  512, 13, 13, 1);
    convCB(actA, s1[1], ws1[1], actB,  512, 1024, 13, 13, 3);
    convCB(actB, s1[2], ws1[2], actA, 1024,  512, 13, 13, 1);
    convCB(actA, s1[3], ws1[3], actB,  512, 1024, 13, 13, 3);
    convCB(actB, s1[4], ws1[4], actA, 1024,  512, 13, 13, 1);   // x1
    convCB(actA, d1cb,  wd1cb,  actB,  512, 1024, 13, 13, 3);
    convDet(actB, wd1, d1_bias, actD, 1024, 13, 13);
    decode_kernel<<<(Bn * 3 * 169 + 255) / 256, 256, 0, stream>>>(
        actD, out, Bn, 13, 0, 32.f, 116.f, 90.f, 156.f, 198.f, 373.f, 326.f);
    convCB(actA, u1, wu1, actB, 512, 256, 13, 13, 1);           // u1(x1)
    {
        const long long tot = (long long)Bn * 26 * 26 * 768;
        upcat_cl16<<<(int)((tot + 255) / 256), 256, 0, stream>>>(actB, f26, actC,
                                                                 Bn, 256, 512, 26, 26);
    }

    // ================= scale 2 : 26x26 =================
    convCB(actC, s2[0], ws2[0], actA, 768, 256, 26, 26, 1);
    convCB(actA, s2[1], ws2[1], actB, 256, 512, 26, 26, 3);
    convCB(actB, s2[2], ws2[2], actA, 512, 256, 26, 26, 1);
    convCB(actA, s2[3], ws2[3], actB, 256, 512, 26, 26, 3);
    convCB(actB, s2[4], ws2[4], actA, 512, 256, 26, 26, 1);     // x2
    convCB(actA, d2cb,  wd2cb,  actB, 256, 512, 26, 26, 3);
    convDet(actB, wd2, d2_bias, actD, 512, 26, 26);
    decode_kernel<<<(Bn * 3 * 676 + 255) / 256, 256, 0, stream>>>(
        actD, out, Bn, 26, 507, 16.f, 30.f, 61.f, 62.f, 45.f, 59.f, 119.f);
    convCB(actA, u2, wu2, actB, 256, 128, 26, 26, 1);           // u2(x2)
    {
        const long long tot = (long long)Bn * 52 * 52 * 384;
        upcat_cl16<<<(int)((tot + 255) / 256), 256, 0, stream>>>(actB, f52, actC,
                                                                 Bn, 128, 256, 52, 52);
    }

    // ================= scale 3 : 52x52 =================
    convCB(actC, s3[0], ws3[0], actA, 384, 128, 52, 52, 1);
    convCB(actA, s3[1], ws3[1], actB, 128, 256, 52, 52, 3);
    convCB(actB, s3[2], ws3[2], actA, 256, 128, 52, 52, 1);
    convCB(actA, s3[3], ws3[3], actB, 128, 256, 52, 52, 3);
    convCB(actB, s3[4], ws3[4], actA, 256, 128, 52, 52, 1);     // x3
    convCB(actA, d3cb,  wd3cb,  actB, 128, 256, 52, 52, 3);
    convDet(actB, wd3, d3_bias, actD, 256, 52, 52);
    decode_kernel<<<(Bn * 3 * 2704 + 255) / 256, 256, 0, stream>>>(
        actD, out, Bn, 52, 2535, 8.f, 10.f, 13.f, 16.f, 30.f, 33.f, 23.f);
}
